// GPTModel_64390149701811
// MI455X (gfx1250) — matrix operand order repaired and checked by reading, not compiled, not run on hardware
//
#include <hip/hip_runtime.h>
#include <hip/hip_bf16.h>
#include <math.h>

// Model dims
#define L_   6
#define H_   12
#define C_   768
#define T_   1024
#define HD_  64
#define B_   2
#define V_   50257
#define BT_  2048
#define C4_  3072
#define VPAD_ 50304   // ceil(V/64)*64
#define BH_  24

typedef __bf16 bf16_t;
typedef __attribute__((ext_vector_type(16))) __bf16 v16bf;
typedef __attribute__((ext_vector_type(8)))  __bf16 v8bf;
typedef __attribute__((ext_vector_type(8)))  float  v8f;

// ---------------------------------------------------------------------------
// WMMA fragment helpers (wave32; A and B share this code's contiguous k order)
// A 16x32 bf16: lane<16 -> row M=lane, K=0..15 contiguous
//               lane>=16 -> row M=lane-16, K=16..31 contiguous
// B 32x16 bf16 from N-major Bt[n][k]: lane<16 -> col N=lane, K=0..15 contiguous
//               lane>=16 -> col N=lane-16, K=16..31 contiguous
// C/D 16x16 f32: vgpr r -> M = r + (lane>=16 ? 8 : 0), N = lane&15
// ---------------------------------------------------------------------------
__device__ __forceinline__ v16bf load_a_frag(const bf16_t* __restrict__ A, int lda,
                                             int m0, int k0, int lane) {
  const int row  = m0 + (lane & 15);
  const int koff = (lane & 16) ? 16 : 0;
  const bf16_t* p = A + (long)row * lda + k0 + koff;
  const v8bf lo = *(const v8bf*)(p);
  const v8bf hi = *(const v8bf*)(p + 8);
  v16bf f;
#pragma unroll
  for (int i = 0; i < 8; ++i) { f[i] = lo[i]; f[i + 8] = hi[i]; }
  return f;
}

__device__ __forceinline__ v16bf load_b_frag(const bf16_t* __restrict__ Bt, int ldb,
                                             int n0, int k0, int lane) {
  const int row  = n0 + (lane & 15);
  const int koff = (lane & 16) ? 16 : 0;
  return *(const v16bf*)(Bt + (long)row * ldb + k0 + koff);
}

__device__ __forceinline__ v8f wmma_bf16(v16bf a, v16bf b, v8f c) {
  return __builtin_amdgcn_wmma_f32_16x16x32_bf16(false, a, false, b, (short)0, c,
                                                 false, false);
}

__device__ __forceinline__ void zero_acc8(v8f acc[2][4]) {
#pragma unroll
  for (int s = 0; s < 2; ++s)
#pragma unroll
    for (int j = 0; j < 4; ++j)
#pragma unroll
      for (int r = 0; r < 8; ++r) acc[s][j][r] = 0.0f;
}

// ---------------------------------------------------------------------------
// Embedding: x[b,t,:] = tok_emb[tokens[b,t]] + pos_emb[t]
// ---------------------------------------------------------------------------
__global__ void k_embed(const int* __restrict__ tok, const float* __restrict__ te,
                        const float* __restrict__ pe, float* __restrict__ x) {
  long i = (long)blockIdx.x * 256 + threadIdx.x;
  if (i >= (long)BT_ * C_) return;
  int c = (int)(i % C_);
  int m = (int)(i / C_);
  int t = m & (T_ - 1);
  x[i] = te[(long)tok[m] * C_ + c] + pe[(long)t * C_ + c];
}

// ---------------------------------------------------------------------------
// Weight conversion: Wq/Wk/Wv [L,H,C,HD] fp32 -> [L][n=h*64+d][k=c] bf16
// ---------------------------------------------------------------------------
__global__ void k_cvt_qkv(const float* __restrict__ src, bf16_t* __restrict__ dst) {
  long i = (long)blockIdx.x * 256 + threadIdx.x;
  if (i >= (long)L_ * C_ * C_) return;
  int cin = (int)(i % C_);
  long r  = i / C_;
  int n   = (int)(r % C_);
  int l   = (int)(r / C_);
  int h = n >> 6, d = n & 63;
  dst[i] = (bf16_t)src[(((long)l * H_ + h) * C_ + cin) * HD_ + d];
}

// Generic transpose-convert: src [Lb][K][N] fp32 -> dst [Lb][Npad][K] bf16 (zero pad)
__global__ void k_cvt_t(const float* __restrict__ src, bf16_t* __restrict__ dst,
                        int K, int N, int Npad, int Lb) {
  long i = (long)blockIdx.x * 256 + threadIdx.x;
  long per = (long)Npad * K;
  if (i >= per * Lb) return;
  int  l  = (int)(i / per);
  long r  = i % per;
  int  kk = (int)(r % K);
  int  n  = (int)(r / K);
  dst[i] = (n < N) ? (bf16_t)src[(long)l * K * N + (long)kk * N + n] : (bf16_t)0.0f;
}

// ---------------------------------------------------------------------------
// LayerNorm (fp32): one workgroup per row; writes fp32 + bf16 copies
// ---------------------------------------------------------------------------
__global__ void __launch_bounds__(256) k_ln(const float* __restrict__ x,
                                            const float* __restrict__ g,
                                            const float* __restrict__ b,
                                            float* __restrict__ outF,
                                            bf16_t* __restrict__ outB) {
  const int row = blockIdx.x;
  const int tid = threadIdx.x;
  const float* xr = x + (long)row * C_;
  __shared__ float red[256];

  float s = 0.0f;
  for (int i = tid; i < C_; i += 256) s += xr[i];
  red[tid] = s; __syncthreads();
  for (int o = 128; o > 0; o >>= 1) {
    if (tid < o) red[tid] += red[tid + o];
    __syncthreads();
  }
  const float mean = red[0] * (1.0f / C_);
  __syncthreads();

  float v = 0.0f;
  for (int i = tid; i < C_; i += 256) { float d = xr[i] - mean; v += d * d; }
  red[tid] = v; __syncthreads();
  for (int o = 128; o > 0; o >>= 1) {
    if (tid < o) red[tid] += red[tid + o];
    __syncthreads();
  }
  const float inv = rsqrtf(red[0] * (1.0f / C_) + 1e-5f);

  for (int i = tid; i < C_; i += 256) {
    float y = (xr[i] - mean) * inv * g[i] + b[i];
    outF[(long)row * C_ + i] = y;
    outB[(long)row * C_ + i] = (bf16_t)y;
  }
}

// ---------------------------------------------------------------------------
// Generic WMMA GEMM: D[m,n] = A[m,:]·Bt[n,:] + bias + resid, opt GELU.
// A bf16 [M,K] row-major, Bt bf16 [Npad,K] N-major.
// Wave computes 32x64 tile (2 M-subtiles x 4 N-subtiles, 8 accumulators):
// per K-step 12 b128 loads feed 8 WMMAs (B fragments amortized over 2 A tiles).
// ---------------------------------------------------------------------------
__global__ void __launch_bounds__(128) k_gemm(const bf16_t* __restrict__ A,
                                              const bf16_t* __restrict__ Bt,
                                              int M, int N, int K,
                                              const float* __restrict__ bias,
                                              const float* __restrict__ resid,
                                              float* __restrict__ outF,
                                              bf16_t* __restrict__ outB,
                                              int gelu_flag) {
  const int lane = threadIdx.x & 31;
  const int wave = threadIdx.x >> 5;
  const int m0 = blockIdx.y * 128 + wave * 32;
  const int n0 = blockIdx.x * 64;
  if (m0 >= M) return;

  v8f acc[2][4];
  zero_acc8(acc);
  for (int k0 = 0; k0 < K; k0 += 32) {
    v16bf a0 = load_a_frag(A, K, m0, k0, lane);
    v16bf a1 = load_a_frag(A, K, m0 + 16, k0, lane);
#pragma unroll
    for (int j = 0; j < 4; ++j) {
      v16bf bb = load_b_frag(Bt, K, n0 + 16 * j, k0, lane);
      acc[0][j] = wmma_bf16(a0, bb, acc[0][j]);
      acc[1][j] = wmma_bf16(a1, bb, acc[1][j]);
    }
  }

#pragma unroll
  for (int s = 0; s < 2; ++s) {
    const int mrow = m0 + 16 * s + ((lane & 16) >> 1);  // +8 for upper half-lanes
#pragma unroll
    for (int j = 0; j < 4; ++j) {
      const int n = n0 + 16 * j + (lane & 15);
      if (n >= N) continue;
#pragma unroll
      for (int r = 0; r < 8; ++r) {
        const int m = mrow + r;
        float v = acc[s][j][r];
        if (bias)  v += bias[n];
        if (resid) v += resid[(long)m * N + n];
        if (gelu_flag) v = 0.5f * v * (1.0f + erff(v * 0.70710678118654752f));
        if (outF) outF[(long)m * N + n] = v;
        if (outB) outB[(long)m * N + n] = (bf16_t)v;
      }
    }
  }
}

// ---------------------------------------------------------------------------
// Fused QKV GEMM (z selects q/k/v weight); q,k stored [B,H,T,HD], v transposed
// to vT [B,H,HD,T] so the A·V GEMM's B-fragments are contiguous.
// ---------------------------------------------------------------------------
__global__ void __launch_bounds__(128) k_gemm_qkv(const bf16_t* __restrict__ A,
                                                  const bf16_t* __restrict__ Wqt,
                                                  const bf16_t* __restrict__ Wkt,
                                                  const bf16_t* __restrict__ Wvt,
                                                  bf16_t* __restrict__ q,
                                                  bf16_t* __restrict__ kmat,
                                                  bf16_t* __restrict__ vT) {
  const int lane = threadIdx.x & 31;
  const int wave = threadIdx.x >> 5;
  const int m0 = blockIdx.y * 128 + wave * 32;
  const int n0 = blockIdx.x * 64;
  const int which = blockIdx.z;
  const bf16_t* Bt = (which == 0) ? Wqt : (which == 1) ? Wkt : Wvt;

  v8f acc[2][4];
  zero_acc8(acc);
  for (int k0 = 0; k0 < C_; k0 += 32) {
    v16bf a0 = load_a_frag(A, C_, m0, k0, lane);
    v16bf a1 = load_a_frag(A, C_, m0 + 16, k0, lane);
#pragma unroll
    for (int j = 0; j < 4; ++j) {
      v16bf bb = load_b_frag(Bt, C_, n0 + 16 * j, k0, lane);
      acc[0][j] = wmma_bf16(a0, bb, acc[0][j]);
      acc[1][j] = wmma_bf16(a1, bb, acc[1][j]);
    }
  }

#pragma unroll
  for (int s = 0; s < 2; ++s) {
    const int mrow = m0 + 16 * s + ((lane & 16) >> 1);
#pragma unroll
    for (int j = 0; j < 4; ++j) {
      const int n = n0 + 16 * j + (lane & 15);
      const int h = n >> 6, d = n & 63;
#pragma unroll
      for (int r = 0; r < 8; ++r) {
        const int m = mrow + r;
        const int bb = m >> 10, t = m & (T_ - 1);
        const bf16_t val = (bf16_t)acc[s][j][r];
        if (which == 2)
          vT[(((long)bb * H_ + h) * HD_ + d) * T_ + t] = val;
        else {
          bf16_t* dst = (which == 0) ? q : kmat;
          dst[(((long)bb * H_ + h) * T_ + t) * HD_ + d] = val;
        }
      }
    }
  }
}

// ---------------------------------------------------------------------------
// Scores: S = Q·K^T per (b,h). k row-major [T,HD] IS the N-major B operand.
// Fully-masked 32x64 wave tiles are skipped (causal).
// ---------------------------------------------------------------------------
__global__ void __launch_bounds__(128) k_scores(const bf16_t* __restrict__ q,
                                                const bf16_t* __restrict__ kmat,
                                                bf16_t* __restrict__ s) {
  const int lane = threadIdx.x & 31;
  const int wave = threadIdx.x >> 5;
  const int m0 = blockIdx.y * 128 + wave * 32;
  const int n0 = blockIdx.x * 64;
  if (n0 > m0 + 31) return;  // entire tile above diagonal
  const long bh = blockIdx.z;
  const bf16_t* A  = q    + bh * (long)T_ * HD_;
  const bf16_t* Bt = kmat + bh * (long)T_ * HD_;

  v8f acc[2][4];
  zero_acc8(acc);
#pragma unroll
  for (int k0 = 0; k0 < HD_; k0 += 32) {
    v16bf a0 = load_a_frag(A, HD_, m0, k0, lane);
    v16bf a1 = load_a_frag(A, HD_, m0 + 16, k0, lane);
#pragma unroll
    for (int j = 0; j < 4; ++j) {
      v16bf bb = load_b_frag(Bt, HD_, n0 + 16 * j, k0, lane);
      acc[0][j] = wmma_bf16(a0, bb, acc[0][j]);
      acc[1][j] = wmma_bf16(a1, bb, acc[1][j]);
    }
  }

#pragma unroll
  for (int ss = 0; ss < 2; ++ss) {
    const int mrow = m0 + 16 * ss + ((lane & 16) >> 1);
#pragma unroll
    for (int j = 0; j < 4; ++j) {
      const int n = n0 + 16 * j + (lane & 15);
#pragma unroll
      for (int r = 0; r < 8; ++r) {
        const int m = mrow + r;
        s[(bh * T_ + m) * (long)T_ + n] = (bf16_t)acc[ss][j][r];
      }
    }
  }
}

// ---------------------------------------------------------------------------
// Causal softmax over one row (b,h,t); scale = C^-0.5 applied here.
// Also zero-fills (t, zend) so the A·V GEMM can run 32-aligned K ranges.
// ---------------------------------------------------------------------------
__global__ void __launch_bounds__(256) k_softmax(bf16_t* __restrict__ s) {
  const int t   = blockIdx.x;
  const long bh = blockIdx.y;
  const int tid = threadIdx.x;
  bf16_t* row = s + (bh * T_ + t) * (long)T_;
  const int len  = t + 1;
  const int zend = ((t >> 5) + 1) << 5;
  const float scale = 0.03608439182435161f;  // 768^-0.5
  __shared__ float red[256];

  float mx = -3.0e38f;
  for (int i = tid; i < len; i += 256) mx = fmaxf(mx, (float)row[i] * scale);
  red[tid] = mx; __syncthreads();
  for (int o = 128; o > 0; o >>= 1) {
    if (tid < o) red[tid] = fmaxf(red[tid], red[tid + o]);
    __syncthreads();
  }
  mx = red[0]; __syncthreads();

  float sum = 0.0f;
  for (int i = tid; i < len; i += 256) sum += __expf((float)row[i] * scale - mx);
  red[tid] = sum; __syncthreads();
  for (int o = 128; o > 0; o >>= 1) {
    if (tid < o) red[tid] += red[tid + o];
    __syncthreads();
  }
  const float inv = 1.0f / red[0];

  for (int i = tid; i < len; i += 256)
    row[i] = (bf16_t)(__expf((float)row[i] * scale - mx) * inv);
  for (int i = len + tid; i < zend; i += 256) row[i] = (bf16_t)0.0f;
}

// ---------------------------------------------------------------------------
// Y = P·V per (b,h); K range limited by causal structure (m0 is 32-aligned, so
// klen = m0+32 == softmax zero-fill boundary); writes head-merged bf16 yb.
// ---------------------------------------------------------------------------
__global__ void __launch_bounds__(128) k_av(const bf16_t* __restrict__ p,
                                            const bf16_t* __restrict__ vT,
                                            bf16_t* __restrict__ yb) {
  const int lane = threadIdx.x & 31;
  const int wave = threadIdx.x >> 5;
  const int m0 = blockIdx.y * 128 + wave * 32;
  const long bh = blockIdx.z;
  const int bb = (int)(bh / H_), h = (int)(bh % H_);
  const bf16_t* A  = p  + bh * (long)T_ * T_;
  const bf16_t* Bt = vT + bh * (long)HD_ * T_;
  const int klen = m0 + 32;

  v8f acc[2][4];
  zero_acc8(acc);
  for (int k0 = 0; k0 < klen; k0 += 32) {
    v16bf a0 = load_a_frag(A, T_, m0, k0, lane);
    v16bf a1 = load_a_frag(A, T_, m0 + 16, k0, lane);
#pragma unroll
    for (int j = 0; j < 4; ++j) {
      v16bf bbf = load_b_frag(Bt, T_, 16 * j, k0, lane);
      acc[0][j] = wmma_bf16(a0, bbf, acc[0][j]);
      acc[1][j] = wmma_bf16(a1, bbf, acc[1][j]);
    }
  }

#pragma unroll
  for (int ss = 0; ss < 2; ++ss) {
    const int mrow = m0 + 16 * ss + ((lane & 16) >> 1);
#pragma unroll
    for (int j = 0; j < 4; ++j) {
      const int n = 16 * j + (lane & 15);  // head dim 0..63
#pragma unroll
      for (int r = 0; r < 8; ++r) {
        const int t = mrow + r;
        yb[((long)bb * T_ + t) * C_ + h * HD_ + n] = (bf16_t)acc[ss][j][r];
      }
    }
  }
}

// ---------------------------------------------------------------------------
extern "C" void kernel_launch(void* const* d_in, const int* in_sizes, int n_in,
                              void* d_out, int out_size, void* d_ws, size_t ws_size,
                              hipStream_t stream) {
  const int*   tokens  = (const int*)  d_in[0];
  const float* tok_emb = (const float*)d_in[1];
  const float* pos_emb = (const float*)d_in[2];
  const float* Wq = (const float*)d_in[3];
  const float* Wk = (const float*)d_in[4];
  const float* Wv = (const float*)d_in[5];
  const float* Wo = (const float*)d_in[6];
  const float* bo = (const float*)d_in[7];
  const float* ln1_g = (const float*)d_in[8];
  const float* ln1_b = (const float*)d_in[9];
  const float* ln2_g = (const float*)d_in[10];
  const float* ln2_b = (const float*)d_in[11];
  const float* W1 = (const float*)d_in[12];
  const float* b1 = (const float*)d_in[13];
  const float* W2 = (const float*)d_in[14];
  const float* b2 = (const float*)d_in[15];
  const float* lnf_g = (const float*)d_in[16];
  const float* lnf_b = (const float*)d_in[17];
  const float* head_w = (const float*)d_in[18];
  const float* head_b = (const float*)d_in[19];
  float* out = (float*)d_out;

  char* ws = (char*)d_ws;
  size_t off = 0;
  auto alloc = [&](size_t bytes) -> char* {
    char* pp = ws + off;
    off = (off + bytes + 255) & ~(size_t)255;
    return pp;
  };

  bf16_t* wqt = (bf16_t*)alloc((size_t)L_ * C_ * C_ * 2);
  bf16_t* wkt = (bf16_t*)alloc((size_t)L_ * C_ * C_ * 2);
  bf16_t* wvt = (bf16_t*)alloc((size_t)L_ * C_ * C_ * 2);
  bf16_t* wot = (bf16_t*)alloc((size_t)L_ * C_ * C_ * 2);
  bf16_t* w1t = (bf16_t*)alloc((size_t)L_ * C4_ * C_ * 2);
  bf16_t* w2t = (bf16_t*)alloc((size_t)L_ * C_ * C4_ * 2);
  bf16_t* ht  = (bf16_t*)alloc((size_t)VPAD_ * C_ * 2);
  float*  x   = (float*) alloc((size_t)BT_ * C_ * 4);
  float*  xn  = (float*) alloc((size_t)BT_ * C_ * 4);
  bf16_t* xnb = (bf16_t*)alloc((size_t)BT_ * C_ * 2);
  bf16_t* qb  = (bf16_t*)alloc((size_t)BT_ * C_ * 2);
  bf16_t* kb  = (bf16_t*)alloc((size_t)BT_ * C_ * 2);
  bf16_t* vTb = (bf16_t*)alloc((size_t)BT_ * C_ * 2);
  bf16_t* sc  = (bf16_t*)alloc((size_t)BH_ * T_ * T_ * 2);
  bf16_t* yb  = (bf16_t*)alloc((size_t)BT_ * C_ * 2);
  bf16_t* hb  = (bf16_t*)alloc((size_t)BT_ * C4_ * 2);

  auto blocks = [](long n) { return (unsigned)((n + 255) / 256); };

  // Weight conversions (stateless: redone every call)
  k_cvt_qkv<<<blocks((long)L_ * C_ * C_), 256, 0, stream>>>(Wq, wqt);
  k_cvt_qkv<<<blocks((long)L_ * C_ * C_), 256, 0, stream>>>(Wk, wkt);
  k_cvt_qkv<<<blocks((long)L_ * C_ * C_), 256, 0, stream>>>(Wv, wvt);
  k_cvt_t<<<blocks((long)L_ * C_ * C_), 256, 0, stream>>>(Wo, wot, C_, C_, C_, L_);
  k_cvt_t<<<blocks((long)L_ * C4_ * C_), 256, 0, stream>>>(W1, w1t, C_, C4_, C4_, L_);
  k_cvt_t<<<blocks((long)L_ * C_ * C4_), 256, 0, stream>>>(W2, w2t, C4_, C_, C_, L_);
  k_cvt_t<<<blocks((long)VPAD_ * C_), 256, 0, stream>>>(head_w, ht, C_, V_, VPAD_, 1);

  // Embedding
  k_embed<<<blocks((long)BT_ * C_), 256, 0, stream>>>(tokens, tok_emb, pos_emb, x);

  for (int l = 0; l < L_; ++l) {
    // ln1 (residual base is the NORMED x, per reference)
    k_ln<<<BT_, 256, 0, stream>>>(x, ln1_g + (long)l * C_, ln1_b + (long)l * C_, xn, xnb);
    // QKV
    k_gemm_qkv<<<dim3(C_ / 64, BT_ / 128, 3), 128, 0, stream>>>(
        xnb, wqt + (long)l * C_ * C_, wkt + (long)l * C_ * C_,
        wvt + (long)l * C_ * C_, qb, kb, vTb);
    // Attention
    k_scores<<<dim3(T_ / 64, T_ / 128, BH_), 128, 0, stream>>>(qb, kb, sc);
    k_softmax<<<dim3(T_, BH_), 256, 0, stream>>>(sc);
    k_av<<<dim3(1, T_ / 128, BH_), 128, 0, stream>>>(sc, vTb, yb);
    // x = xn + y@Wo + bo
    k_gemm<<<dim3(C_ / 64, BT_ / 128), 128, 0, stream>>>(
        yb, wot + (long)l * C_ * C_, BT_, C_, C_,
        bo + (long)l * C_, xn, x, nullptr, 0);
    // ln2 (residual base for MLP is normed x, per reference)
    k_ln<<<BT_, 256, 0, stream>>>(x, ln2_g + (long)l * C_, ln2_b + (long)l * C_, xn, xnb);
    // h = gelu(xn@W1 + b1)
    k_gemm<<<dim3(C4_ / 64, BT_ / 128), 128, 0, stream>>>(
        xnb, w1t + (long)l * C4_ * C_, BT_, C4_, C_,
        b1 + (long)l * C4_, nullptr, nullptr, hb, 1);
    // x = xn + h@W2 + b2
    k_gemm<<<dim3(C_ / 64, BT_ / 128), 128, 0, stream>>>(
        hb, w2t + (long)l * C_ * C4_, BT_, C_, C4_,
        b2 + (long)l * C_, xn, x, nullptr, 0);
  }

  // Final LN + LM head
  k_ln<<<BT_, 256, 0, stream>>>(x, lnf_g, lnf_b, xn, xnb);
  k_gemm<<<dim3(VPAD_ / 64, BT_ / 128), 128, 0, stream>>>(
      xnb, ht, BT_, V_, C_, head_b, nullptr, out, nullptr, 0);
}
